// SpikingLinear_ev_20933670601080
// MI455X (gfx1250) — compile-verified, matrix operand order checked by reading, not run on hardware
//
#include <hip/hip_runtime.h>

// ---------------- problem constants (from reference) ----------------
#define STEPS 100
#define BATCH 256
#define D1    784
#define D2    512
#define KPAD  800                     // 784 padded to multiple of 32
#define MROWS ((STEPS - 1) * BATCH)   // 25344 = rows of the big GEMM (uses x[:-1])
#define NN    (BATCH * D2)            // 131072 neurons

#define ALPHA_C 0.8f                  // 1 - dt/tau_s
#define BETA_C  0.95f                 // 1 - dt/tau_m
#define ONE_MINUS_BETA 0.05f

typedef _Float16 v16h __attribute__((ext_vector_type(16)));
typedef _Float16 v8h  __attribute__((ext_vector_type(8)));
typedef float    v8f  __attribute__((ext_vector_type(8)));

// ---------------- f32 -> f16 conversion with K padding ----------------
// 2D grid: y = row, x covers KPAD columns. Pads k in [D1, KPAD) with zero.
__global__ void cvt_pad_f16(const float* __restrict__ src, _Float16* __restrict__ dst) {
    int k = blockIdx.x * blockDim.x + threadIdx.x;
    int m = blockIdx.y;
    if (k >= KPAD) return;
    _Float16 v = (_Float16)0.0f;
    if (k < D1) v = (_Float16)src[(long)m * D1 + k];
    dst[(long)m * KPAD + k] = v;
}

// ---------------- WMMA GEMM: C[M,N] = A[M,K] * B[N,K]^T ----------------
// A: [MROWS, KPAD] f16 row-major (spikes, t*batch flattened)
// B: [D2, KPAD]    f16 row-major (weight layout already N x K)
// C: [MROWS, D2]   f32
// Each wave: 2 M-tiles x 4 N-tiles of 16x16 (32x64 macro-tile), K in steps of 32.
__global__ __launch_bounds__(256)
void gemm_wmma_f16(const _Float16* __restrict__ A,
                   const _Float16* __restrict__ B,
                   float* __restrict__ C) {
    const int wid  = (blockIdx.x * blockDim.x + threadIdx.x) >> 5;  // global wave id
    const int lane = threadIdx.x & 31;

    const int m2 = wid >> 3;        // 792 M positions (32 rows each)
    const int n4 = wid & 7;         // 8   N positions (64 cols each)
    const int m0 = m2 * 32;
    const int n0 = n4 * 64;

    // ISA 7.12.2 f16 A/B fragment layout (wave32):
    //   lane L (<16): row = L,  K halves {kb..kb+7} U {kb+16..kb+23}, kb = 0
    //   lane L (>=16): same row L-16's pair with kb = 8
    const int row = lane & 15;
    const int kb  = (lane >> 4) * 8;

    const _Float16* aBase = A + (long)(m0 + row) * KPAD + kb;
    const _Float16* bBase = B + (long)(n0 + row) * KPAD + kb;

    v8f acc[2][4] = {};

    for (int k0 = 0; k0 < KPAD; k0 += 32) {
        v16h af[2];
        v16h bf[4];
#pragma unroll
        for (int i = 0; i < 2; ++i) {
            const _Float16* p = aBase + (long)i * 16 * KPAD + k0;
            v8h lo = *(const v8h*)(p);        // K = k0+kb   .. +7   (16B aligned)
            v8h hi = *(const v8h*)(p + 16);   // K = k0+kb+16 .. +23
            af[i] = __builtin_shufflevector(lo, hi, 0,1,2,3,4,5,6,7,
                                                   8,9,10,11,12,13,14,15);
        }
#pragma unroll
        for (int j = 0; j < 4; ++j) {
            const _Float16* p = bBase + (long)j * 16 * KPAD + k0;
            v8h lo = *(const v8h*)(p);
            v8h hi = *(const v8h*)(p + 16);
            bf[j] = __builtin_shufflevector(lo, hi, 0,1,2,3,4,5,6,7,
                                                   8,9,10,11,12,13,14,15);
        }
#pragma unroll
        for (int i = 0; i < 2; ++i)
#pragma unroll
            for (int j = 0; j < 4; ++j)
                acc[i][j] = __builtin_amdgcn_wmma_f32_16x16x32_f16(
                    /*neg_a=*/false, af[i],
                    /*neg_b=*/false, bf[j],
                    /*c_mod=*/(short)0, acc[i][j],
                    /*reuse_a=*/false, /*reuse_b=*/false);
    }

    // C/D layout: VGPR r -> row r (lanes 0-15) or row r+8 (lanes 16-31), col = lane&15
    const int mloc = (lane >> 4) * 8;
    const int nloc = lane & 15;
#pragma unroll
    for (int i = 0; i < 2; ++i)
#pragma unroll
        for (int j = 0; j < 4; ++j) {
            float* cp = C + (long)(m0 + i * 16 + mloc) * D2 + (n0 + j * 16 + nloc);
#pragma unroll
            for (int r = 0; r < 8; ++r)
                cp[(long)r * D2] = acc[i][j][r];
        }
}

// ---------------- LIF recurrence scan (sequential in t, parallel in neurons) ----
__global__ __launch_bounds__(256)
void lif_scan(const float* __restrict__ C, float* __restrict__ out) {
    const int idx = blockIdx.x * blockDim.x + threadIdx.x;   // [0, NN)
    float I = 0.0f, V = 0.0f;
    out[idx] = 0.0f;                                         // spikes[0] = 0
#pragma unroll 1
    for (int t = 0; t < STEPS - 1; ++t) {
        const float c  = C[(long)t * NN + idx];
        I              = ALPHA_C * I + c;
        const float vp = BETA_C * V + ONE_MINUS_BETA * I;
        const float s  = (vp > 1.0f) ? 1.0f : 0.0f;
        out[(long)(t + 1) * NN + idx] = s;
        V = (1.0f - s) * vp;
    }
}

// ---------------- host launcher ----------------
extern "C" void kernel_launch(void* const* d_in, const int* in_sizes, int n_in,
                              void* d_out, int out_size, void* d_ws, size_t ws_size,
                              hipStream_t stream) {
    const float* x = (const float*)d_in[0];   // [STEPS, BATCH, D1] f32 (0/1 spikes)
    const float* w = (const float*)d_in[1];   // [D2, D1] f32
    float* out = (float*)d_out;               // [STEPS, BATCH, D2] f32

    // workspace layout (bytes):
    //   A_f16: MROWS*KPAD*2 = 40,550,400
    //   B_f16: D2*KPAD*2    =    819,200
    //   C_f32: MROWS*D2*4   = 51,904,512
    char* ws = (char*)d_ws;
    _Float16* A16 = (_Float16*)(ws);
    _Float16* B16 = (_Float16*)(ws + (size_t)MROWS * KPAD * 2);
    float*    Cbuf = (float*)(ws + (size_t)MROWS * KPAD * 2 + (size_t)D2 * KPAD * 2);

    // 1) convert + pad spikes (only first STEPS-1 time steps are consumed)
    {
        dim3 grid((KPAD + 255) / 256, MROWS);
        cvt_pad_f16<<<grid, 256, 0, stream>>>(x, A16);
    }
    // 2) convert + pad weights
    {
        dim3 grid((KPAD + 255) / 256, D2);
        cvt_pad_f16<<<grid, 256, 0, stream>>>(w, B16);
    }
    // 3) big WMMA GEMM: 792 M-tiles(32) x 8 N-tiles(64) = 6336 waves = 792 blocks
    {
        const int waves = (MROWS / 32) * (D2 / 64);
        gemm_wmma_f16<<<waves / 8, 256, 0, stream>>>(A16, B16, Cbuf);
    }
    // 4) LIF recurrence: one thread per neuron
    {
        lif_scan<<<NN / 256, 256, 0, stream>>>(Cbuf, out);
    }
}
